// SP_DNN_61125974557283
// MI455X (gfx1250) — matrix operand order repaired and checked by reading, not compiled, not run on hardware
//
#include <hip/hip_runtime.h>
#include <hip/hip_bf16.h>

typedef _Float16 half_t;
typedef __attribute__((ext_vector_type(16))) _Float16 v16h;
typedef __attribute__((ext_vector_type(8)))  _Float16 v8h;
typedef __attribute__((ext_vector_type(8)))  float    v8f;

#define HDIM 256
#define SROW 264                 // padded LDS row stride in halfs (528B)
#define ROWS_PER_WAVE 16
#define WAVES_PER_WG 2
#define ROWS_PER_WG (ROWS_PER_WAVE * WAVES_PER_WG)
#define KT_TILES 8               // 256 / 32
#define CT_TILES 16              // 256 / 16
#define TILE_HALFS 512           // 32 lanes * 16 halfs per B tile
#define LAYER_HALFS (CT_TILES * KT_TILES * TILE_HALFS)   // 65536 halfs / layer

// ---------------------------------------------------------------------------
// Pack fp32 weights W[k][n] (k=in,n=out) into the WMMA B-matrix lane layout:
//   B (32x16 f16): lane l holds column n=l%16; halves j<8 hold K=8*(l/16)+j, j>=8 hold K=16+8*(l/16)+(j-8)
// Tile-major: [ct*8+kt][lane*16 + j], 6 layers: xW1,xW2,xW3,tW1,tW2,tW3.
// ---------------------------------------------------------------------------
__global__ void pack_weights_k(const float* __restrict__ w0, const float* __restrict__ w1,
                               const float* __restrict__ w2, const float* __restrict__ w3,
                               const float* __restrict__ w4, const float* __restrict__ w5,
                               half_t* __restrict__ out) {
    int tid = blockIdx.x * blockDim.x + threadIdx.x;
    if (tid >= 6 * LAYER_HALFS) return;
    int layer = tid >> 16;
    int rem   = tid & (LAYER_HALFS - 1);
    int tile  = rem >> 9;            // ct*8 + kt
    int kt    = tile & 7;
    int e     = rem & 511;
    int l     = e >> 4;              // lane 0..31
    int j     = e & 15;              // half index 0..15
    int ct    = tile >> 3;
    int k = (j & 7) + ((j >> 3) << 4) + ((l >> 4) << 3) + (kt << 5);      // K index 0..255
    int n = (ct << 4) + (l & 15);                 // N index 0..255
    const float* W = (layer == 0) ? w0 : (layer == 1) ? w1 : (layer == 2) ? w2
                   : (layer == 3) ? w3 : (layer == 4) ? w4 : w5;
    out[tid] = (half_t)W[k * HDIM + n];
}

// ---------------------------------------------------------------------------
// A-tile load (ISA 16-bit A 16x32 layout): lane l -> row m=l%16,
// halves j<8 -> K = 32*kt + 8*(l/16) + j ; halves j>=8 -> +16.
// ---------------------------------------------------------------------------
__device__ __forceinline__ v16h load_a(const half_t* __restrict__ buf, int kt, int lane) {
    int row = lane & 15;
    int k0  = (kt << 5) + ((lane >> 4) << 3);
    v8h lo = *(const v8h*)(buf + row * SROW + k0);
    v8h hi = *(const v8h*)(buf + row * SROW + k0 + 16);
    v16h a;
#pragma unroll
    for (int i = 0; i < 8; ++i) { a[i] = lo[i]; a[i + 8] = hi[i]; }
    return a;
}

__device__ __forceinline__ v16h load_b(const half_t* __restrict__ wp, int ct, int kt, int lane) {
    return *(const v16h*)(wp + (((ct << 3) + kt) << 9) + (lane << 4));
}

// ---------------------------------------------------------------------------
// Layer 0: h[m][n] = sin(x[m]*W0[n] + b0[n]), 16 rows x 256 cols -> LDS f16.
// ---------------------------------------------------------------------------
__device__ __forceinline__ void layer0(const float* __restrict__ xin, int comp,
                                       const float* __restrict__ W0, const float* __restrict__ b0,
                                       half_t* __restrict__ out, int row0, int lane, int Ntot) {
    float w[8], bb[8];
#pragma unroll
    for (int i = 0; i < 8; ++i) { w[i] = W0[lane * 8 + i]; bb[i] = b0[lane * 8 + i]; }
    for (int r = 0; r < ROWS_PER_WAVE; ++r) {
        int row = row0 + r;
        if (row >= Ntot) row = Ntot - 1;           // clamp: keeps EXEC full
        float xv = xin[row * 2 + comp];
        v8h h;
#pragma unroll
        for (int i = 0; i < 8; ++i) h[i] = (half_t)__sinf(xv * w[i] + bb[i]);
        *(v8h*)(out + r * SROW + lane * 8) = h;
    }
}

// ---------------------------------------------------------------------------
// Hidden layer: out = sin(in @ W + b), 16 rows, full 256x256, WMMA f16->f32.
// Accumulators start from inline 0 (wmma C = literal 0); bias added in the
// sin epilogue so no accumulator-broadcast movs sit on the critical path.
// ---------------------------------------------------------------------------
__device__ __forceinline__ void hidden_layer(const half_t* __restrict__ in,
                                             half_t* __restrict__ out,
                                             const half_t* __restrict__ wp,
                                             const float* __restrict__ bias, int lane) {
    v16h A[KT_TILES];
#pragma unroll
    for (int kt = 0; kt < KT_TILES; ++kt) A[kt] = load_a(in, kt, lane);
    int nl = lane & 15;
    int m0 = (lane >> 4) << 3;
    for (int ct = 0; ct < CT_TILES; ++ct) {
        float bv = bias[ct * 16 + nl];
        v8f c = {};
#pragma unroll
        for (int kt = 0; kt < KT_TILES; ++kt) {
            v16h b = load_b(wp, ct, kt, lane);
            c = __builtin_amdgcn_wmma_f32_16x16x32_f16(false, A[kt], false, b,
                                                       (short)0, c, false, false);
        }
#pragma unroll
        for (int r = 0; r < 8; ++r)
            out[(m0 + r) * SROW + ct * 16 + nl] = (half_t)__sinf(c[r] + bv);
    }
}

// ---------------------------------------------------------------------------
// Fused final layers: Dx = hx2@xW3+xb3, Dt = ht2@tW3+tb3 share the same C/D
// lane layout (m = r + 8*(lane/16), n = lane%16 + 16*ct), so the even/odd
// paired dot product is acc[r] += Dx[r]*Dt[r] with parity = lane&1, then a
// lane-xor {2,4,8} reduction.
// ---------------------------------------------------------------------------
__device__ __forceinline__ void last_layer(const half_t* __restrict__ bufx,
                                           const half_t* __restrict__ buft,
                                           const half_t* __restrict__ wpx,
                                           const half_t* __restrict__ wpt,
                                           const float* __restrict__ bx3,
                                           const float* __restrict__ bt3,
                                           float* __restrict__ outp,
                                           int row0, int lane, int Ntot) {
    v16h Ax[KT_TILES], At[KT_TILES];
#pragma unroll
    for (int kt = 0; kt < KT_TILES; ++kt) {
        Ax[kt] = load_a(bufx, kt, lane);
        At[kt] = load_a(buft, kt, lane);
    }
    int nl = lane & 15;
    float acc[8];
#pragma unroll
    for (int r = 0; r < 8; ++r) acc[r] = 0.0f;
    for (int ct = 0; ct < CT_TILES; ++ct) {
        float bvx = bx3[ct * 16 + nl];
        float bvt = bt3[ct * 16 + nl];
        v8f cx = {}, ctv = {};
#pragma unroll
        for (int kt = 0; kt < KT_TILES; ++kt) {
            cx  = __builtin_amdgcn_wmma_f32_16x16x32_f16(false, Ax[kt], false,
                      load_b(wpx, ct, kt, lane), (short)0, cx,  false, false);
            ctv = __builtin_amdgcn_wmma_f32_16x16x32_f16(false, At[kt], false,
                      load_b(wpt, ct, kt, lane), (short)0, ctv, false, false);
        }
#pragma unroll
        for (int r = 0; r < 8; ++r) acc[r] += (cx[r] + bvx) * (ctv[r] + bvt);
    }
#pragma unroll
    for (int r = 0; r < 8; ++r) {
        acc[r] += __shfl_xor(acc[r], 2, 32);
        acc[r] += __shfl_xor(acc[r], 4, 32);
        acc[r] += __shfl_xor(acc[r], 8, 32);
    }
    if ((lane & 14) == 0) {          // lanes 0,1 (m=0..7) and 16,17 (m=8..15)
        int p     = lane & 1;        // output parity column
        int mbase = row0 + ((lane >> 4) << 3);
#pragma unroll
        for (int r = 0; r < 8; ++r) {
            int row = mbase + r;
            if (row < Ntot) outp[row * 2 + p] = acc[r];
        }
    }
}

__global__ void __launch_bounds__(WAVES_PER_WG * 32)
mlp_pairdot_kernel(const float* __restrict__ x,
                   const float* __restrict__ xW0, const float* __restrict__ xb0,
                   const float* __restrict__ xb1, const float* __restrict__ xb2,
                   const float* __restrict__ xb3,
                   const float* __restrict__ tW0, const float* __restrict__ tb0,
                   const float* __restrict__ tb1, const float* __restrict__ tb2,
                   const float* __restrict__ tb3,
                   const half_t* __restrict__ wpack,
                   float* __restrict__ outp, int Ntot) {
    extern __shared__ half_t smem[];
    int lane = threadIdx.x & 31;
    int wave = threadIdx.x >> 5;
    half_t* buf0 = smem + wave * 3 * (ROWS_PER_WAVE * SROW);
    half_t* buf1 = buf0 + ROWS_PER_WAVE * SROW;
    half_t* buf2 = buf1 + ROWS_PER_WAVE * SROW;
    int row0 = blockIdx.x * ROWS_PER_WG + wave * ROWS_PER_WAVE;

    // x-MLP through layer 2 -> hx2 in buf0
    layer0(x, 0, xW0, xb0, buf0, row0, lane, Ntot);
    asm volatile("s_wait_dscnt 0" ::: "memory");
    hidden_layer(buf0, buf1, wpack + 0 * LAYER_HALFS, xb1, lane);
    asm volatile("s_wait_dscnt 0" ::: "memory");
    hidden_layer(buf1, buf0, wpack + 1 * LAYER_HALFS, xb2, lane);
    asm volatile("s_wait_dscnt 0" ::: "memory");

    // t-MLP through layer 2 -> ht2 in buf1
    layer0(x, 1, tW0, tb0, buf1, row0, lane, Ntot);
    asm volatile("s_wait_dscnt 0" ::: "memory");
    hidden_layer(buf1, buf2, wpack + 3 * LAYER_HALFS, tb1, lane);
    asm volatile("s_wait_dscnt 0" ::: "memory");
    hidden_layer(buf2, buf1, wpack + 4 * LAYER_HALFS, tb2, lane);
    asm volatile("s_wait_dscnt 0" ::: "memory");

    // fused final layers + paired even/odd dot product
    last_layer(buf0, buf1, wpack + 2 * LAYER_HALFS, wpack + 5 * LAYER_HALFS,
               xb3, tb3, outp, row0, lane, Ntot);
}

extern "C" void kernel_launch(void* const* d_in, const int* in_sizes, int n_in,
                              void* d_out, int out_size, void* d_ws, size_t ws_size,
                              hipStream_t stream) {
    const float* x   = (const float*)d_in[0];
    const float* xW0 = (const float*)d_in[1];
    const float* xb0 = (const float*)d_in[2];
    const float* xW1 = (const float*)d_in[3];
    const float* xb1 = (const float*)d_in[4];
    const float* xW2 = (const float*)d_in[5];
    const float* xb2 = (const float*)d_in[6];
    const float* xW3 = (const float*)d_in[7];
    const float* xb3 = (const float*)d_in[8];
    const float* tW0 = (const float*)d_in[9];
    const float* tb0 = (const float*)d_in[10];
    const float* tW1 = (const float*)d_in[11];
    const float* tb1 = (const float*)d_in[12];
    const float* tW2 = (const float*)d_in[13];
    const float* tb2 = (const float*)d_in[14];
    const float* tW3 = (const float*)d_in[15];
    const float* tb3 = (const float*)d_in[16];

    int Ntot = in_sizes[0] / 2;
    half_t* wpack = (half_t*)d_ws;            // needs 6*65536*2 = 786432 B

    int total = 6 * LAYER_HALFS;
    pack_weights_k<<<(total + 255) / 256, 256, 0, stream>>>(xW1, xW2, xW3,
                                                            tW1, tW2, tW3, wpack);

    int nblocks  = (Ntot + ROWS_PER_WG - 1) / ROWS_PER_WG;
    size_t ldsB  = (size_t)WAVES_PER_WG * 3 * ROWS_PER_WAVE * SROW * sizeof(half_t);
    mlp_pairdot_kernel<<<nblocks, WAVES_PER_WG * 32, ldsB, stream>>>(
        x, xW0, xb0, xb1, xb2, xb3, tW0, tb0, tb1, tb2, tb3,
        wpack, (float*)d_out, Ntot);
}
